// CNF_48558900248735
// MI455X (gfx1250) — compile-verified, matrix operand order checked by reading, not run on hardware
//
#include <hip/hip_runtime.h>
#include <hip/hip_bf16.h>
#include <cstdint>
#include <cstddef>

// Problem sizes (fixed by the reference)
#define B_SZ 4096
#define D_SZ 64
#define H_SZ 1024

typedef __attribute__((ext_vector_type(16))) __bf16 bf16x16;
typedef __attribute__((ext_vector_type(8)))  float  f32x8;

struct FragBits { uint4 lo, hi; };

// ---- CDNA5 async copy: global -> LDS, 16 bytes per lane, tracked by ASYNCcnt ----
static __device__ __forceinline__ void async_copy_b128(uint32_t lds_byte_off, const void* gaddr) {
    asm volatile("global_load_async_to_lds_b128 %0, %1, off"
                 :: "v"(lds_byte_off), "v"(gaddr)
                 : "memory");
}

static __device__ __forceinline__ void wait_async0() {
#if __has_builtin(__builtin_amdgcn_s_wait_asynccnt)
    __builtin_amdgcn_s_wait_asynccnt(0);
#else
    asm volatile("s_wait_asynccnt 0x0" ::: "memory");
#endif
}

// Load one 16x32 bf16 WMMA fragment from an LDS tile (row stride ldt elements).
// Lane l (m = l&15, half = l>>4): VGPRs 0-3 hold K = 8*half + 0..7,
// VGPRs 4-7 hold K = 16 + 8*half + 0..7  -> two 16-byte ds_load_b128.
static __device__ __forceinline__ bf16x16
lds_frag(const __bf16* __restrict__ tile, int ldt, int r0, int k0, int lane) {
    const int m = lane & 15, half = lane >> 4;
    const __bf16* p = tile + (r0 + m) * ldt + k0 + half * 8;
    FragBits fb;
    fb.lo = *(const uint4*)(p);
    fb.hi = *(const uint4*)(p + 16);
    return __builtin_bit_cast(bf16x16, fb);
}

// Generic bf16 GEMM with fp32 accumulate:  C(MxN) = A(MxK) @ BT(NxK)^T
// Block tile 128 x BN, double-buffered async LDS staging (K-panel depth 64),
// wave tile 32 x (BN/2).
// MODE 0: z = acc + bias[col]; h = tanh(z); outH = bf16(h); outS = bf16(1-h^2)
// MODE 1: outF[row*N + col] = acc                        (G matrix)
// MODE 2: outF[row*65 + 1 + col] = acc + bias[col]       (x_out into d_out)
template <int MODE, int BN>
__global__ __launch_bounds__(256) void gemm_wmma_bf16(
    const __bf16* __restrict__ A,
    const __bf16* __restrict__ BT,
    const float*  __restrict__ bias,
    int M, int N, int K,
    __bf16* __restrict__ outH,
    __bf16* __restrict__ outS,
    float*  __restrict__ outF)
{
    constexpr int BM  = 128;           // block rows
    constexpr int KS  = 64;            // staged K-panel depth (2 WMMA depths)
    constexpr int LDT = 72;            // LDS tile row stride (elems): 64 + 8 pad
    constexpr int FN  = BN / 32;       // B fragments per wave (wave tile 32 x 16*FN)
    constexpr int A_ELEMS = BM * LDT;  // one A buffer
    constexpr int B_ELEMS = BN * LDT;  // one B buffer

    // Single shared object => LDS byte offset 0; async offsets computed from indices.
    __shared__ __align__(16) __bf16 smem[2 * A_ELEMS + 2 * B_ELEMS];

    const int tid  = threadIdx.x;
    const int lane = tid & 31;
    const int wave = tid >> 5;
    const int wm   = wave & 3;         // 4 waves along M
    const int wn   = wave >> 2;        // 2 waves along N
    const int rowB = blockIdx.y * BM;
    const int colB = blockIdx.x * BN;

    f32x8 acc[2][FN];
#pragma unroll
    for (int i = 0; i < 2; ++i)
#pragma unroll
        for (int j = 0; j < FN; ++j)
            acc[i][j] = f32x8{};

    // Per-thread staging geometry (compile-time shaped, constant across k-steps).
    // Panel is (rows x 64 elems) = rows x 8 chunks of 16B; thread tid covers
    // chunk rows (tid>>3) + 32*q at column (tid&7)*8.
    const int      rS = tid >> 3;                     // 0..31
    const int      cS = (tid & 7) * 8;                // 0..56
    const __bf16*  gA = A  + (size_t)(rowB + rS) * K + cS;
    const __bf16*  gB = BT + (size_t)(colB + rS) * K + cS;
    const uint32_t lS = (uint32_t)(rS * LDT + cS) * 2u;   // LDS byte off within buffer
    const size_t   sK = (size_t)32 * (size_t)K;           // 32-row global stride (elems)
    constexpr uint32_t L32 = (uint32_t)(32 * LDT * 2);    // 32-row LDS stride (bytes)

    auto stage = [&](int buf, int k0) {
        const uint32_t bA = (uint32_t)buf * (uint32_t)(A_ELEMS * 2);
        const uint32_t bB = (uint32_t)(2 * A_ELEMS * 2) + (uint32_t)buf * (uint32_t)(B_ELEMS * 2);
#pragma unroll
        for (int q = 0; q < BM / 32; ++q)
            async_copy_b128(bA + lS + (uint32_t)q * L32, gA + (size_t)k0 + (size_t)q * sK);
#pragma unroll
        for (int q = 0; q < BN / 32; ++q)
            async_copy_b128(bB + lS + (uint32_t)q * L32, gB + (size_t)k0 + (size_t)q * sK);
    };

    const int nk = K / KS;
    stage(0, 0);
    wait_async0();
    __syncthreads();

    for (int i = 0; i < nk; ++i) {
        const int buf = i & 1;
        if (i + 1 < nk) stage(buf ^ 1, (i + 1) * KS);

        const __bf16* At = &smem[buf * A_ELEMS];
        const __bf16* Bt = &smem[2 * A_ELEMS + buf * B_ELEMS];

#pragma unroll
        for (int ks = 0; ks < KS; ks += 32) {
            bf16x16 a0 = lds_frag(At, LDT, wm * 32,      ks, lane);
            bf16x16 a1 = lds_frag(At, LDT, wm * 32 + 16, ks, lane);
#pragma unroll
            for (int fn = 0; fn < FN; ++fn) {
                bf16x16 b = lds_frag(Bt, LDT, wn * (16 * FN) + fn * 16, ks, lane);
                acc[0][fn] = __builtin_amdgcn_wmma_f32_16x16x32_bf16(false, a0, false, b, (short)0, acc[0][fn], false, false);
                acc[1][fn] = __builtin_amdgcn_wmma_f32_16x16x32_bf16(false, a1, false, b, (short)0, acc[1][fn], false, false);
            }
        }

        wait_async0();       // staged buffer complete (this wave's share)
        __syncthreads();     // all waves done reading `buf` and staging `buf^1`
    }

    const int half = lane >> 4;
    const int n    = lane & 15;
#pragma unroll
    for (int fm = 0; fm < 2; ++fm) {
#pragma unroll
        for (int fn = 0; fn < FN; ++fn) {
#pragma unroll
            for (int r = 0; r < 8; ++r) {
                const int row = rowB + wm * 32 + fm * 16 + half * 8 + r;
                const int col = colB + wn * (16 * FN) + fn * 16 + n;
                const float v = acc[fm][fn][r];
                if (MODE == 0) {
                    const float z = v + bias[col];
                    const float h = tanhf(z);
                    outH[(size_t)row * N + col] = (__bf16)h;
                    outS[(size_t)row * N + col] = (__bf16)(1.0f - h * h);
                } else if (MODE == 1) {
                    outF[(size_t)row * N + col] = v;
                } else {
                    outF[(size_t)row * 65 + 1 + col] = v + bias[col];
                }
            }
        }
    }
}

// dst (C x R, bf16) = transpose of src (R x C, f32)
__global__ void k_transpose_to_bf16(__bf16* __restrict__ dst,
                                    const float* __restrict__ src,
                                    int R, int C) {
    const int i = blockIdx.x * blockDim.x + threadIdx.x;
    if (i >= R * C) return;
    const int c = i / R;
    const int r = i - c * R;
    dst[i] = (__bf16)src[(size_t)r * C + c];
}

// Xbf[b][d] = bf16( x[b][1+d] )
__global__ void k_cvt_x(__bf16* __restrict__ Xbf, const float* __restrict__ x) {
    const int i = blockIdx.x * blockDim.x + threadIdx.x;
    if (i >= B_SZ * D_SZ) return;
    const int b = i / D_SZ;
    const int d = i - b * D_SZ;
    Xbf[i] = (__bf16)x[(size_t)b * (1 + D_SZ) + 1 + d];
}

// QT[j][k] = ( sum_i W3[j][i] * W1[i][k] ) * W2[k][j]   (row-major N=j, K=k)
__global__ __launch_bounds__(256) void k_build_qt(__bf16* __restrict__ QT,
                                                  const float* __restrict__ W1,
                                                  const float* __restrict__ W2,
                                                  const float* __restrict__ W3) {
    const int k = blockIdx.x * 16 + (threadIdx.x & 15);
    const int j = blockIdx.y * 16 + (threadIdx.x >> 4);
    float acc = 0.0f;
#pragma unroll 4
    for (int i = 0; i < D_SZ; ++i)
        acc += W3[(size_t)j * D_SZ + i] * W1[(size_t)i * H_SZ + k];
    QT[(size_t)j * H_SZ + k] = (__bf16)(acc * W2[(size_t)k * H_SZ + j]);
}

// out[b][0] = - sum_j s2[b][j] * G[b][j]   (one wave32 per row)
__global__ __launch_bounds__(256) void k_trj(const __bf16* __restrict__ s2,
                                             const float*  __restrict__ G,
                                             float* __restrict__ out) {
    const int lane = threadIdx.x & 31;
    const int wave = threadIdx.x >> 5;
    const int row  = blockIdx.x * 8 + wave;
    float acc = 0.0f;
    for (int j = lane; j < H_SZ; j += 32)
        acc += (float)s2[(size_t)row * H_SZ + j] * G[(size_t)row * H_SZ + j];
#pragma unroll
    for (int off = 16; off > 0; off >>= 1)
        acc += __shfl_down(acc, off, 32);
    if (lane == 0) out[(size_t)row * 65] = -acc;
}

extern "C" void kernel_launch(void* const* d_in, const int* in_sizes, int n_in,
                              void* d_out, int out_size, void* d_ws, size_t ws_size,
                              hipStream_t stream) {
    (void)in_sizes; (void)n_in; (void)out_size; (void)ws_size;

    const float* x  = (const float*)d_in[0];
    const float* W1 = (const float*)d_in[1];
    const float* b1 = (const float*)d_in[2];
    const float* W2 = (const float*)d_in[3];
    const float* b2 = (const float*)d_in[4];
    const float* W3 = (const float*)d_in[5];
    const float* b3 = (const float*)d_in[6];
    float* out = (float*)d_out;

    char* ws = (char*)d_ws;
    size_t off = 0;
    auto alloc = [&](size_t bytes) -> void* {
        void* p = ws + off;
        off += (bytes + 255) & ~(size_t)255;
        return p;
    };

    __bf16* W1T = (__bf16*)alloc((size_t)H_SZ * D_SZ * 2);  // H x D
    __bf16* W2T = (__bf16*)alloc((size_t)H_SZ * H_SZ * 2);  // H x H
    __bf16* W3T = (__bf16*)alloc((size_t)D_SZ * H_SZ * 2);  // D x H
    __bf16* QT  = (__bf16*)alloc((size_t)H_SZ * H_SZ * 2);  // H x H
    __bf16* Xbf = (__bf16*)alloc((size_t)B_SZ * D_SZ * 2);  // B x D
    __bf16* h1  = (__bf16*)alloc((size_t)B_SZ * H_SZ * 2);
    __bf16* s1  = (__bf16*)alloc((size_t)B_SZ * H_SZ * 2);
    __bf16* h2  = (__bf16*)alloc((size_t)B_SZ * H_SZ * 2);
    __bf16* s2  = (__bf16*)alloc((size_t)B_SZ * H_SZ * 2);
    float*  G   = (float*) alloc((size_t)B_SZ * H_SZ * 4);

    // --- preprocessing: bf16 transposed weights, bf16 input slice, Q^T ---
    k_transpose_to_bf16<<<(H_SZ * D_SZ + 255) / 256, 256, 0, stream>>>(W1T, W1, D_SZ, H_SZ);
    k_transpose_to_bf16<<<(H_SZ * H_SZ + 255) / 256, 256, 0, stream>>>(W2T, W2, H_SZ, H_SZ);
    k_transpose_to_bf16<<<(D_SZ * H_SZ + 255) / 256, 256, 0, stream>>>(W3T, W3, H_SZ, D_SZ);
    k_cvt_x<<<(B_SZ * D_SZ + 255) / 256, 256, 0, stream>>>(Xbf, x);
    k_build_qt<<<dim3(H_SZ / 16, H_SZ / 16), 256, 0, stream>>>(QT, W1, W2, W3);

    // --- layer 1: h1 = tanh(x @ W1 + b1), s1 = 1 - h1^2 ---
    gemm_wmma_bf16<0, 128><<<dim3(H_SZ / 128, B_SZ / 128), 256, 0, stream>>>(
        Xbf, W1T, b1, B_SZ, H_SZ, D_SZ, h1, s1, nullptr);

    // --- layer 2: h2 = tanh(h1 @ W2 + b2), s2 = 1 - h2^2 ---
    gemm_wmma_bf16<0, 128><<<dim3(H_SZ / 128, B_SZ / 128), 256, 0, stream>>>(
        h1, W2T, b2, B_SZ, H_SZ, H_SZ, h2, s2, nullptr);

    // --- trace GEMM: G = s1 @ Q  (Q given as QT, N x K) ---
    gemm_wmma_bf16<1, 128><<<dim3(H_SZ / 128, B_SZ / 128), 256, 0, stream>>>(
        s1, QT, nullptr, B_SZ, H_SZ, H_SZ, nullptr, nullptr, G);

    // --- layer 3: x_out = h2 @ W3 + b3 -> out[:, 1:65] ---
    gemm_wmma_bf16<2, 64><<<dim3(D_SZ / 64, B_SZ / 128), 256, 0, stream>>>(
        h2, W3T, b3, B_SZ, D_SZ, H_SZ, nullptr, nullptr, out);

    // --- out[:, 0] = -rowdot(s2, G) ---
    k_trj<<<B_SZ / 8, 256, 0, stream>>>(s2, G, out);
}
